// AlloBiCTCLayer_7653631722156
// MI455X (gfx1250) — matrix operand order repaired and checked by reading, not compiled, not run on hardware
//
#include <hip/hip_runtime.h>

// ---------------------------------------------------------------------------
// AlloBiCTC on MI455X (gfx1250, wave32)
//   K1: allo softmax fp32->fp16                  [P=256][C=512]      (tiny)
//   K2: hs softmax + f16 WMMA GEMM + log -> em   [B*T=32768][P=256]  (HBM bound)
//   K3: CTC bigram forward scan (seq over T)     one WG per batch
//   K4: ordered mean reduction -> d_out          (deterministic, no fp atomics)
// ---------------------------------------------------------------------------

typedef __attribute__((ext_vector_type(16))) _Float16 v16h;
typedef __attribute__((ext_vector_type(8)))  _Float16 v8h;
typedef __attribute__((ext_vector_type(4)))  _Float16 v4h;
typedef __attribute__((ext_vector_type(8)))  float    v8f;

constexpr int BB = 32, TT = 1024, CC = 512, PP = 256, LL = 128;
constexpr int SS = 2 * LL + 1;           // 257 lattice states
constexpr float NEGINF = -1e30f;

// ---------------- K1: softmax over allophone weights, store fp16 row-major --
__global__ void k_allo_softmax(const float* __restrict__ W,
                               _Float16* __restrict__ alloH) {
    const int p = blockIdx.x;            // one wave per phoneme row
    const int lane = threadIdx.x;        // 32 threads
    const float* row = W + (size_t)p * CC;
    float v[16];
    float mx = -INFINITY;
#pragma unroll
    for (int i = 0; i < 16; ++i) { v[i] = row[i * 32 + lane]; mx = fmaxf(mx, v[i]); }
#pragma unroll
    for (int off = 16; off > 0; off >>= 1) mx = fmaxf(mx, __shfl_xor(mx, off, 32));
    float sum = 0.f;
#pragma unroll
    for (int i = 0; i < 16; ++i) { v[i] = __expf(v[i] - mx); sum += v[i]; }
#pragma unroll
    for (int off = 16; off > 0; off >>= 1) sum += __shfl_xor(sum, off, 32);
    const float inv = 1.0f / sum;
#pragma unroll
    for (int i = 0; i < 16; ++i)
        alloH[(size_t)p * CC + i * 32 + lane] = (_Float16)(v[i] * inv);
}

// ---------------- K2: fused softmax(hs) @ softmax(allo)^T via WMMA ----------
// Block: 256 threads (8 waves) -> 16 (b,t) rows, full P=256 output.
// Wave w: softmax rows 2w,2w+1 into LDS (f16), then 2 N-tiles (n0=32w,+16),
// K swept 0..511 in steps of 32 with v_wmma_f32_16x16x32_f16.
__global__ void __launch_bounds__(256)
k_emissions(const float* __restrict__ hs, const _Float16* __restrict__ alloH,
            float* __restrict__ em) {
    __shared__ __align__(32) _Float16 probs[16 * CC];   // 16 KB
    const int tid  = threadIdx.x;
    const int wave = tid >> 5;
    const int lane = tid & 31;
    const long r0  = (long)blockIdx.x * 16;             // first (b,t) row

    // ---- phase 1: row softmax (fp32 in regs) -> fp16 probs in LDS ----
#pragma unroll
    for (int rr = 0; rr < 2; ++rr) {
        const int row = wave * 2 + rr;
        const float4* src4 = (const float4*)(hs + (r0 + row) * (long)CC);
        float4 q[4];
        float mx = -INFINITY;
#pragma unroll
        for (int j = 0; j < 4; ++j) {
            q[j] = src4[j * 32 + lane];                 // coalesced b128
            mx = fmaxf(mx, fmaxf(fmaxf(q[j].x, q[j].y), fmaxf(q[j].z, q[j].w)));
        }
#pragma unroll
        for (int off = 16; off > 0; off >>= 1) mx = fmaxf(mx, __shfl_xor(mx, off, 32));
        float sum = 0.f;
#pragma unroll
        for (int j = 0; j < 4; ++j) {
            q[j].x = __expf(q[j].x - mx); q[j].y = __expf(q[j].y - mx);
            q[j].z = __expf(q[j].z - mx); q[j].w = __expf(q[j].w - mx);
            sum += q[j].x + q[j].y + q[j].z + q[j].w;
        }
#pragma unroll
        for (int off = 16; off > 0; off >>= 1) sum += __shfl_xor(sum, off, 32);
        const float inv = 1.0f / sum;
#pragma unroll
        for (int j = 0; j < 4; ++j) {
            v4h h;
            h[0] = (_Float16)(q[j].x * inv); h[1] = (_Float16)(q[j].y * inv);
            h[2] = (_Float16)(q[j].z * inv); h[3] = (_Float16)(q[j].w * inv);
            *(v4h*)&probs[row * CC + (j * 32 + lane) * 4] = h;
        }
    }
    __syncthreads();

    // ---- phase 2: WMMA over K=512, two 16-wide N tiles per wave ----
    const int m  = lane & 15;            // M row (A) / N col (B,D) within tile
    const int hi = lane >> 4;            // lane-half selector
    const int n0 = wave * 32;

    v8f c0 = {};
    v8f c1 = {};
    for (int k0 = 0; k0 < CC; k0 += 32) {
        // A 16x32 f16 layout: lane half holds K = k0+16*hi+[0..15] (contig, matches B)
        v8h a_lo = *(const v8h*)&probs[m * CC + k0 + 16 * hi];
        v8h a_hi = *(const v8h*)&probs[m * CC + k0 + 16 * hi + 8];
        v16h a = __builtin_shufflevector(a_lo, a_hi,
                 0,1,2,3,4,5,6,7,8,9,10,11,12,13,14,15);
        // B 32x16 f16: lane half holds K = k0+16*hi+[0..15] of column n (contig)
        v16h b0 = *(const v16h*)&alloH[(size_t)(n0 + m)      * CC + k0 + 16 * hi];
        v16h b1 = *(const v16h*)&alloH[(size_t)(n0 + 16 + m) * CC + k0 + 16 * hi];
        c0 = __builtin_amdgcn_wmma_f32_16x16x32_f16(false, a, false, b0,
                                                    (short)0, c0, false, false);
        c1 = __builtin_amdgcn_wmma_f32_16x16x32_f16(false, a, false, b1,
                                                    (short)0, c1, false, false);
    }

    // ---- phase 3: log-clamp and store (D: VGPR r -> M=r+8*hi, N=lane&15) ----
#pragma unroll
    for (int r = 0; r < 8; ++r) {
        const long rowg = r0 + r + 8 * hi;
        em[rowg * PP + n0 + m]      = __logf(fmaxf(c0[r], 1e-30f));
        em[rowg * PP + n0 + 16 + m] = __logf(fmaxf(c1[r], 1e-30f));
    }
}

// ---------------- K3: CTC bigram forward scan -------------------------------
// One block per batch element; thread s owns lattice state s (s < 257).
// Double-buffered alpha in LDS, one barrier per timestep, emission gather
// pipelined one step ahead to hide global latency behind the exp/log chain.
__global__ void __launch_bounds__(288)
k_ctc_scan(const float* __restrict__ em, const float* __restrict__ biph,
           const int* __restrict__ targets, float* __restrict__ partial) {
    const int b = blockIdx.x;
    const int s = threadIdx.x;
    __shared__ int   ext[SS];
    __shared__ float A[2][SS + 3];
    __shared__ float res2[2];

    if (s < SS) ext[s] = (s & 1) ? targets[b * LL + (s >> 1)] : 0;
    __syncthreads();

    const float* emb = em + (size_t)b * TT * PP;
    int   e = 0;
    float tstay = 0.f, tprev = 0.f, tskip = 0.f;
    bool  skipok = false;
    if (s < SS) {
        e = ext[s];
        const int ep1 = (s >= 1) ? ext[s - 1] : ext[0];
        const int ep2 = (s >= 2) ? ext[s - 2] : ext[s];
        tstay = biph[e * PP + e];
        tprev = biph[ep1 * PP + e];
        tskip = biph[ep2 * PP + e];
        skipok = ((s & 1) == 1) && (s >= 2) && (e != ep2);
        A[0][s] = (s <= 1) ? emb[e] : NEGINF;          // alpha0 = em[b,0,ext[s]]
    }
    __syncthreads();

    float em_t = (s < SS) ? emb[(size_t)PP + e] : 0.f; // emission for t=1
    int pb = 0;
    for (int t = 1; t < TT; ++t) {
        float em_n = 0.f;
        if (t + 1 < TT && s < SS) em_n = emb[(size_t)(t + 1) * PP + e];
        if (s < SS) {
            const float a_st = A[pb][s] + tstay;
            const float a_pr = ((s >= 1) ? A[pb][s - 1] : NEGINF) + tprev;
            const float a_sk = skipok ? (A[pb][s - 2] + tskip) : NEGINF;
            const float mm = fmaxf(a_st, fmaxf(a_pr, a_sk));
            A[pb ^ 1][s] = em_t + mm +
                __logf(__expf(a_st - mm) + __expf(a_pr - mm) + __expf(a_sk - mm));
        }
        __syncthreads();
        pb ^= 1;
        em_t = em_n;
    }

    if (s == SS - 1 || s == SS - 2) res2[s - (SS - 2)] = A[pb][s];
    __syncthreads();
    if (s == 0) {
        const float x = res2[0], y = res2[1];
        const float mm = fmaxf(x, y);
        partial[b] = -(mm + __logf(__expf(x - mm) + __expf(y - mm)));
    }
}

// ---------------- K4: ordered mean over batch (deterministic) ---------------
__global__ void k_reduce(const float* __restrict__ partial, float* __restrict__ out) {
    float sum = 0.f;
    for (int i = 0; i < BB; ++i) sum += partial[i];
    out[0] = sum / (float)BB;
}

// ---------------------------------------------------------------------------
extern "C" void kernel_launch(void* const* d_in, const int* in_sizes, int n_in,
                              void* d_out, int out_size, void* d_ws, size_t ws_size,
                              hipStream_t stream) {
    const float* hs      = (const float*)d_in[0];   // [B,T,C] fp32
    const float* alloW   = (const float*)d_in[1];   // [P,C]   fp32
    const float* biph    = (const float*)d_in[2];   // [P,P]   fp32
    const int*   targets = (const int*)d_in[3];     // [B,L]   int32
    float*       out     = (float*)d_out;           // scalar loss

    char* ws = (char*)d_ws;
    const size_t alloBytes = (size_t)PP * CC * sizeof(_Float16);   // 256 KB
    const size_t emBytes   = (size_t)BB * TT * PP * sizeof(float); // 32 MB
    _Float16* alloH   = (_Float16*)ws;
    float*    em      = (float*)(ws + alloBytes);
    float*    partial = (float*)(ws + alloBytes + emBytes);

    k_allo_softmax<<<dim3(PP), dim3(32), 0, stream>>>(alloW, alloH);
    k_emissions<<<dim3(BB * TT / 16), dim3(256), 0, stream>>>(hs, alloH, em);
    k_ctc_scan<<<dim3(BB), dim3(288), 0, stream>>>(em, biph, targets, partial);
    k_reduce<<<dim3(1), dim3(1), 0, stream>>>(partial, out);
}
